// GatedSelfAttention_44959717655286
// MI455X (gfx1250) — matrix operand order repaired and checked by reading, not compiled, not run on hardware
//
#include <hip/hip_runtime.h>

// ---------------------------------------------------------------------------
// GatedSelfAttention for MI455X (gfx1250): bf16 WMMA + async LDS staging.
// B=2, L=1024, D=2048, H=16, HKV=4, HD=128, CACHE=1024, S=2048.
// Softmax runs max-free: RMS-normed q,k with 128^-0.5 scaling bound
// |score| <= sqrt(128) ~= 11.31, so exp never overflows fp32.
// ---------------------------------------------------------------------------

typedef unsigned short u16_t;
typedef unsigned int   u32_t;
typedef __attribute__((ext_vector_type(16))) __bf16 v16bf;
typedef __attribute__((ext_vector_type(8)))  __bf16 v8bf;
typedef __attribute__((ext_vector_type(2)))  __bf16 v2bf;
typedef __attribute__((ext_vector_type(8)))  float  v8f;
typedef __attribute__((ext_vector_type(4)))  int    v4i;

union V8F  { v8f  v; float f[8]; };
union V16B { v16bf v; u16_t u[16]; };
union PK2  { v2bf  v; u32_t u; u16_t h[2]; };

#define BB     2
#define LL     1024
#define DD     2048
#define HH     16
#define HKV    4
#define REP    4
#define HD     128
#define CACHE  1024
#define SS     2048
#define BL     (BB * LL)      // 2048
#define NKV    (HKV * HD)     // 512
// 128^-0.5 * log2(e): scores become log2-domain for native v_exp_f32
#define QK_SCALE_LOG2E (0.08838834764831845f * 1.4426950408889634f)

// ---- CDNA5 async global->LDS path (ASYNCcnt), guarded for portability ----
#if defined(__AMDGCN__) && __has_builtin(__builtin_amdgcn_global_load_async_to_lds_b128) && __has_builtin(__builtin_amdgcn_s_wait_asynccnt)
#define ASYNC_LDS 1
#else
#define ASYNC_LDS 0
#endif

__device__ __forceinline__ void cp16_g2l(const u16_t* g, u16_t* l) {
#if ASYNC_LDS
  __builtin_amdgcn_global_load_async_to_lds_b128(
      (__attribute__((address_space(1))) v4i*)((void*)g),
      (__attribute__((address_space(3))) v4i*)((void*)l), 0, 0);
#else
  *(uint4*)l = *(const uint4*)g;
#endif
}
__device__ __forceinline__ void async_wait0() {
#if ASYNC_LDS
  __builtin_amdgcn_s_wait_asynccnt(0);
#endif
}

#if defined(__AMDGCN__) && __has_builtin(__builtin_amdgcn_exp2f)
#define fast_exp2(x) __builtin_amdgcn_exp2f(x)
#else
#define fast_exp2(x) exp2f(x)
#endif

__device__ __forceinline__ u16_t f2bf(float x) {
  union { float f; u32_t u; } c; c.f = x;
  u32_t lsb = (c.u >> 16) & 1u;
  return (u16_t)((c.u + 0x7fffu + lsb) >> 16);
}

// Two f32 -> packed 2x bf16 in one-to-three VALU ops.
__device__ __forceinline__ u32_t pk_bf16(float a, float b) {
#if defined(__AMDGCN__) && __has_builtin(__builtin_amdgcn_cvt_pk_bf16_f32)
  PK2 p; p.v = __builtin_amdgcn_cvt_pk_bf16_f32(a, b);
  return p.u;
#elif defined(__AMDGCN__) && __has_builtin(__builtin_amdgcn_perm)
  // round-half-up then v_perm_b32 packs the two high halves: {b_hi16, a_hi16}
  union { float f; u32_t u; } ca, cb; ca.f = a; cb.f = b;
  return __builtin_amdgcn_perm(cb.u + 0x8000u, ca.u + 0x8000u, 0x07060302u);
#else
  return (u32_t)f2bf(a) | ((u32_t)f2bf(b) << 16);
#endif
}

// CDNA5 16-bit A fragment (16x32 MxK): lane holds row (lane&15);
// K chunks [ (lane>>4)*8 , +8 ) and [ 16+(lane>>4)*8 , +8 ).
__device__ __forceinline__ v16bf load_a_frag(const u16_t* row, int cA) {
  union { v16bf v; v8bf h[2]; } u;
  u.h[0] = *(const v8bf*)(row + cA * 8);
  u.h[1] = *(const v8bf*)(row + 16 + cA * 8);
  return u.v;
}
// CDNA5 16-bit B fragment (32x16 KxN): lane holds column (lane&15);
// K chunks [ (lane>>4)*8 , +8 ) and [ 16+(lane>>4)*8 , +8 ).
__device__ __forceinline__ v16bf load_b_frag(const u16_t* row, int cA) {
  union { v16bf v; v8bf h[2]; } u;
  u.h[0] = *(const v8bf*)(row + cA * 8);
  u.h[1] = *(const v8bf*)(row + 16 + cA * 8);
  return u.v;
}

__device__ __forceinline__ v8f wmma_bf16(v16bf a, v16bf b, v8f c) {
  return __builtin_amdgcn_wmma_f32_16x16x32_bf16(false, a, false, b,
                                                 (short)0, c, false, false);
}

// ------------------------------ converters ---------------------------------
__global__ __launch_bounds__(256) void f32_to_bf16_vec4(const float* __restrict__ src,
                                                        u16_t* __restrict__ dst, int n4) {
  int i = blockIdx.x * 256 + threadIdx.x;
  if (i >= n4) return;
  float4 v = ((const float4*)src)[i];
  uint2 o;
  o.x = pk_bf16(v.x, v.y);
  o.y = pk_bf16(v.z, v.w);
  ((uint2*)dst)[i] = o;
}

// cache_k [B,HKV,CACHE,HD] f32 -> Kt [B*HKV, S, HD] bf16 at s in [0,CACHE)
__global__ __launch_bounds__(256) void convert_cache_k(const float* __restrict__ ck,
                                                       u16_t* __restrict__ Kt, int n2) {
  int i = blockIdx.x * 256 + threadIdx.x;      // processes 2 elems along hd
  if (i >= n2) return;
  int hd = (i & (HD / 2 - 1)) * 2;
  int s  = (i >> 6) & (CACHE - 1);
  int kv = i >> 16;                            // b*HKV + kh
  const float* p = ck + (size_t)i * 2;
  *(u32_t*)(Kt + ((size_t)kv * SS + s) * HD + hd) = pk_bf16(p[0], p[1]);
}

// cache_v [B,HKV,CACHE,HD] f32 -> Vt [B*HKV, HD, S] bf16 (transposed)
__global__ __launch_bounds__(256) void convert_cache_v(const float* __restrict__ cv,
                                                       u16_t* __restrict__ Vt, int n) {
  int i = blockIdx.x * 256 + threadIdx.x;
  if (i >= n) return;
  int hd = i & (HD - 1);
  int s  = (i >> 7) & (CACHE - 1);
  int kv = i >> 17;
  Vt[((size_t)kv * HD + hd) * SS + s] = f2bf(cv[i]);
}

// Yv f32 [BL, NKV] -> Vt [B*HKV, HD, S] bf16 at s = CACHE + l
__global__ __launch_bounds__(256) void scatter_v(const float* __restrict__ Yv,
                                                 u16_t* __restrict__ Vt, int n) {
  int i = blockIdx.x * 256 + threadIdx.x;
  if (i >= n) return;
  int c = i & (NKV - 1);  int m = i >> 9;
  int hd = c & (HD - 1);  int kh = c >> 7;
  int l = m & (LL - 1);   int b = m >> 10;
  Vt[(((size_t)(b * HKV + kh)) * HD + hd) * SS + CACHE + l] = f2bf(Yv[i]);
}

// ------------------------------ WMMA GEMM ----------------------------------
// C[M,N] (f32) = A[M,K] bf16 row-major * B[N,K]^T bf16 row-major.
// Block 256 thr = 8 waves; block tile 128x128; wave tile 32x64; K-step 32.
// A/B K-slices double-buffered in LDS via async global->LDS copies.
#define GP 40                 // LDS pitch (halfwords) for 32-wide K slices
template <int N, int K>
__global__ __launch_bounds__(256) void gemm_bf16_nt(const u16_t* __restrict__ A,
                                                    const u16_t* __restrict__ B,
                                                    float* __restrict__ C) {
  __shared__ __align__(16) u16_t shA[2][128 * GP];
  __shared__ __align__(16) u16_t shB[2][128 * GP];
  const int tid  = threadIdx.x;
  const int lane = tid & 31;
  const int wv   = tid >> 5;
  const int wm   = wv & 3, wn = wv >> 2;
  const int gm   = blockIdx.y * 128;
  const int gn   = blockIdx.x * 128;
  const int rA   = lane & 15, cA = lane >> 4;

  V8F acc[2][4];
#pragma unroll
  for (int i = 0; i < 2; ++i)
#pragma unroll
    for (int j = 0; j < 4; ++j)
#pragma unroll
      for (int v = 0; v < 8; ++v) acc[i][j].f[v] = 0.f;

  auto stage = [&](int buf, int k0) {
#pragma unroll
    for (int ci = 0; ci < 4; ++ci) {          // 1024 16B chunks / 256 threads
      int c   = tid + ci * 256;
      int isB = c >> 9;
      int cc  = c & 511;
      int row = cc >> 2;                      // 0..127
      int sub = cc & 3;                       // 16B chunk within 64B row
      const u16_t* g = (isB ? B + (size_t)(gn + row) * K
                            : A + (size_t)(gm + row) * K) + k0 + sub * 8;
      u16_t* l = (isB ? shB[buf] : shA[buf]) + row * GP + sub * 8;
      cp16_g2l(g, l);
    }
  };

  stage(0, 0);
  const int nsteps = K >> 5;
  for (int s = 0; s < nsteps; ++s) {
    const int cur = s & 1;
    async_wait0();
    __syncthreads();
    if (s + 1 < nsteps) stage(cur ^ 1, (s + 1) << 5);
    v16bf af0 = load_a_frag(shA[cur] + (wm * 32 + rA) * GP, cA);
    v16bf af1 = load_a_frag(shA[cur] + (wm * 32 + 16 + rA) * GP, cA);
#pragma unroll
    for (int j = 0; j < 4; ++j) {
      v16bf bf = load_b_frag(shB[cur] + (wn * 64 + j * 16 + rA) * GP, cA);
      acc[0][j].v = wmma_bf16(af0, bf, acc[0][j].v);
      acc[1][j].v = wmma_bf16(af1, bf, acc[1][j].v);
    }
  }
  // C/D layout: lane holds D[v + 8*(lane>>4)][lane&15] in vgpr v.
#pragma unroll
  for (int i = 0; i < 2; ++i)
#pragma unroll
    for (int j = 0; j < 4; ++j) {
      int row = gm + wm * 32 + 16 * i + 8 * cA;
      int col = gn + wn * 64 + 16 * j + rA;
      float* cp = C + (size_t)row * N + col;
#pragma unroll
      for (int v = 0; v < 8; ++v) cp[(size_t)v * N] = acc[i][j].f[v];
    }
}

// --------------------------- RMS-norm epilogues ----------------------------
__global__ __launch_bounds__(256) void rmsnorm_q(const float* __restrict__ Yq,
                                                 u16_t* __restrict__ Qb) {
  int lane = threadIdx.x & 31, wv = threadIdx.x >> 5;
  int row = blockIdx.x * 8 + wv;            // row = m*16 + h, m = b*L + l
  int h = row & 15, m = row >> 4;
  int l = m & (LL - 1), b = m >> 10;
  const float* src = Yq + (size_t)m * DD + h * HD + lane * 4;
  float4 xv = *(const float4*)src;
  float ss = xv.x * xv.x + xv.y * xv.y + xv.z * xv.z + xv.w * xv.w;
#pragma unroll
  for (int off = 16; off >= 1; off >>= 1) ss += __shfl_xor(ss, off, 32);
  // fold qk scale AND log2(e): scores land in log2 domain for v_exp_f32
  float r = rsqrtf(ss * (1.f / 128.f) + 1e-6f) * QK_SCALE_LOG2E;
  int kh = h >> 2, rr = h & 3;
  size_t d = ((((size_t)b * HKV + kh) * REP + rr) * LL + l) * HD + lane * 4;
  uint2 st;
  st.x = pk_bf16(xv.x * r, xv.y * r);
  st.y = pk_bf16(xv.z * r, xv.w * r);
  *(uint2*)(Qb + d) = st;
}

__global__ __launch_bounds__(256) void rmsnorm_k(const float* __restrict__ Yk,
                                                 u16_t* __restrict__ Kt) {
  int lane = threadIdx.x & 31, wv = threadIdx.x >> 5;
  int row = blockIdx.x * 8 + wv;            // row = m*4 + kh
  int kh = row & 3, m = row >> 2;
  int l = m & (LL - 1), b = m >> 10;
  const float* src = Yk + (size_t)m * NKV + kh * HD + lane * 4;
  float4 xv = *(const float4*)src;
  float ss = xv.x * xv.x + xv.y * xv.y + xv.z * xv.z + xv.w * xv.w;
#pragma unroll
  for (int off = 16; off >= 1; off >>= 1) ss += __shfl_xor(ss, off, 32);
  float r = rsqrtf(ss * (1.f / 128.f) + 1e-6f);
  size_t d = (((size_t)(b * HKV + kh)) * SS + CACHE + l) * HD + lane * 4;
  uint2 st;
  st.x = pk_bf16(xv.x * r, xv.y * r);
  st.y = pk_bf16(xv.z * r, xv.w * r);
  *(uint2*)(Kt + d) = st;
}

// ------------------------------- gating ------------------------------------
__global__ __launch_bounds__(256) void gates_kernel(const float* __restrict__ x,
                                                    const float* __restrict__ wg,
                                                    float* __restrict__ gates) {
  int i = blockIdx.x * 256 + threadIdx.x;   // i = (b*L + l)*16 + h
  int h = i & 15, m = i >> 4;
  const float* xr = x + (size_t)m * DD;
  const float* wr = wg + h * HH;
  float acc = 0.f;
#pragma unroll
  for (int j = 0; j < HH; ++j) acc += xr[j] * wr[j];
  gates[i] = 1.f / (1.f + __expf(-acc));
}

// --------------------------- flash attention -------------------------------
// Block = 8 waves = 8 consecutive 16-query tiles sharing ONE KV head.
// K/V 32-key chunks double-buffered in LDS via async copies (8x VMEM reuse).
// Max-free softmax: p = exp2(score_log2); row sums via WMMA with all-ones B
// (each D column equals the row sum -> replicated across the 16-lane halves,
// exactly the layout the normalization needs). No cross-lane shuffles at all.
#define KP 136                // LDS pitch for K rows (128 hd + pad)
#define VP 40                 // LDS pitch for V rows (32 keys + pad)
#define PP 40                 // LDS pitch for P tile rows
__global__ __launch_bounds__(256) void attn_kernel(const u16_t* __restrict__ Qb,
                                                   const u16_t* __restrict__ Kt,
                                                   const u16_t* __restrict__ Vt,
                                                   const float* __restrict__ gates,
                                                   u16_t* __restrict__ Og) {
  __shared__ __align__(16) u16_t shK[2][32 * KP];    // [key][hd]
  __shared__ __align__(16) u16_t shV[2][128 * VP];   // [hd][key]
  __shared__ __align__(16) u16_t plds[8][16 * PP];
  const int tid  = threadIdx.x;
  const int lane = tid & 31;
  const int wv   = tid >> 5;
  const int tile = blockIdx.x * 8 + wv;     // 0..2047; 8 tiles share bhr
  const int lt   = tile & 63;               // L/16 tiles
  const int bhr  = tile >> 6;               // (b*HKV + kh)*REP + r (block-uniform)
  const int kv   = bhr >> 2;                // b*HKV + kh
  const int l0   = lt * 16;
  const int rA = lane & 15, cA = lane >> 4;

  const u16_t* qbase = Qb + ((size_t)bhr * LL + l0) * HD;
  const u16_t* kbase = Kt + (size_t)kv * SS * HD;
  const u16_t* vbase = Vt + (size_t)kv * HD * SS;
  u16_t* pl = plds[wv];

  auto stageKV = [&](int buf, int s0) {
#pragma unroll
    for (int ci = 0; ci < 4; ++ci) {        // 1024 16B chunks / 256 threads
      int c   = tid + ci * 256;
      int isV = c >> 9;
      int cc  = c & 511;
      if (!isV) {
        int row = cc >> 4;                  // key 0..31
        int sub = cc & 15;                  // 16B chunk of 256B row
        cp16_g2l(kbase + (size_t)(s0 + row) * HD + sub * 8,
                 shK[buf] + row * KP + sub * 8);
      } else {
        int row = cc >> 2;                  // hd 0..127
        int sub = cc & 3;
        cp16_g2l(vbase + (size_t)row * SS + s0 + sub * 8,
                 shV[buf] + row * VP + sub * 8);
      }
    }
  };

  // Q tile fragments are reused by every key chunk: hoist out of the S loop.
  v16bf aq[4];
#pragma unroll
  for (int kk = 0; kk < 4; ++kk)
    aq[kk] = load_a_frag(qbase + (size_t)rA * HD + kk * 32, cA);

  // All-ones B fragment for WMMA row sums (bf16 1.0 = 0x3F80).
  V16B ones;
#pragma unroll
  for (int i = 0; i < 16; ++i) ones.u[i] = 0x3F80;

  V8F o[8];                                 // o[t].f[v]: row v+8*cA, hd t*16+rA
  V8F rs;                                   // row sums, replicated over lanes
#pragma unroll
  for (int t = 0; t < 8; ++t)
#pragma unroll
    for (int v = 0; v < 8; ++v) o[t].f[v] = 0.f;
#pragma unroll
  for (int v = 0; v < 8; ++v) rs.f[v] = 0.f;

  stageKV(0, 0);
  for (int s0 = 0; s0 < SS; s0 += 32) {
    const int cur = (s0 >> 5) & 1;
    async_wait0();
    __syncthreads();
    if (s0 + 32 < SS) stageKV(cur ^ 1, s0 + 32);

    // ---- scores (log2 domain): 16 q rows x 32 keys from LDS ----
    V8F sc[2];
#pragma unroll
    for (int j = 0; j < 2; ++j)
#pragma unroll
      for (int v = 0; v < 8; ++v) sc[j].f[v] = 0.f;
#pragma unroll
    for (int kk = 0; kk < 4; ++kk) {
      v16bf k0 = load_b_frag(shK[cur] + (rA) * KP + kk * 32, cA);
      v16bf k1 = load_b_frag(shK[cur] + (16 + rA) * KP + kk * 32, cA);
      sc[0].v = wmma_bf16(aq[kk], k0, sc[0].v);
      sc[1].v = wmma_bf16(aq[kk], k1, sc[1].v);
    }
    // ---- p = exp2(score); |score_ln| <= sqrt(128): no max subtraction ----
#pragma unroll
    for (int v = 0; v < 8; ++v) {
      float p0 = fast_exp2(sc[0].f[v]);
      float p1 = fast_exp2(sc[1].f[v]);
      PK2 w; w.u = pk_bf16(p0, p1);         // v_perm_b32 pack
      int row = v + 8 * cA;
      pl[row * PP + rA]      = w.h[0];
      pl[row * PP + 16 + rA] = w.h[1];
    }
    // ---- O += P*V ; rowsum += P*1 (sum replicated across columns) ----
    v16bf ap = load_a_frag(pl + rA * PP, cA);
    rs.v = wmma_bf16(ap, ones.v, rs.v);
#pragma unroll
    for (int t = 0; t < 8; ++t) {
      v16bf bv = load_b_frag(shV[cur] + (t * 16 + rA) * VP, cA);
      o[t].v = wmma_bf16(ap, bv, o[t].v);
    }
  }

  // ---- epilogue: 1/rowsum, sigmoid gate, store bf16 [BL, D] ----
  int r = bhr & 3, kh = kv & 3, b = kv >> 2;
  int h = kh * REP + r;
#pragma unroll
  for (int v = 0; v < 8; ++v) {
    int lrow = l0 + v + 8 * cA;
    float g = gates[((size_t)b * LL + lrow) * HH + h];
    float inv = g / rs.f[v];
    size_t orow = ((size_t)b * LL + lrow) * DD + h * HD;
#pragma unroll
    for (int t = 0; t < 8; t += 2) {
      PK2 w; w.u = pk_bf16(o[t].f[v] * inv, o[t + 1].f[v] * inv);
      Og[orow + t * 16 + rA]       = w.h[0];
      Og[orow + (t + 1) * 16 + rA] = w.h[1];
    }
  }
}

// ------------------------------- launcher ----------------------------------
extern "C" void kernel_launch(void* const* d_in, const int* in_sizes, int n_in,
                              void* d_out, int out_size, void* d_ws, size_t ws_size,
                              hipStream_t stream) {
  (void)in_sizes; (void)n_in; (void)out_size; (void)ws_size;
  const float* x   = (const float*)d_in[0];
  const float* wq  = (const float*)d_in[1];
  const float* wk  = (const float*)d_in[2];
  const float* wvp = (const float*)d_in[3];
  const float* wo  = (const float*)d_in[4];
  const float* wg  = (const float*)d_in[5];
  const float* ck  = (const float*)d_in[6];
  const float* cv  = (const float*)d_in[7];
  float* out = (float*)d_out;

  char* ws = (char*)d_ws;
  size_t off = 0;
  auto take = [&](size_t bytes) -> char* {
    char* p = ws + off; off += (bytes + 255) & ~(size_t)255; return p;
  };

  u16_t* x_bf  = (u16_t*)take((size_t)BL * DD * 2);
  u16_t* wq_bf = (u16_t*)take((size_t)DD * DD * 2);
  u16_t* wk_bf = (u16_t*)take((size_t)NKV * DD * 2);
  u16_t* wv_bf = (u16_t*)take((size_t)NKV * DD * 2);
  u16_t* wo_bf = (u16_t*)take((size_t)DD * DD * 2);
  float* Yq    = (float*)take((size_t)BL * DD * 4);
  float* Yk    = (float*)take((size_t)BL * NKV * 4);
  float* Yv    = (float*)take((size_t)BL * NKV * 4);
  u16_t* Qb    = (u16_t*)take((size_t)BL * DD * 2);
  u16_t* Kt    = (u16_t*)take((size_t)BB * HKV * SS * HD * 2);
  u16_t* Vt    = (u16_t*)take((size_t)BB * HKV * SS * HD * 2);
  // aliases (stream-ordered reuse after producers finish):
  float* gates = (float*)Yk;                // written after rmsnorm_k
  u16_t* Og    = (u16_t*)Yq;                // written after rmsnorm_q

  const int T = 256;
  f32_to_bf16_vec4<<<(BL * DD / 4 + T - 1) / T, T, 0, stream>>>(x, x_bf, BL * DD / 4);
  f32_to_bf16_vec4<<<(DD * DD / 4 + T - 1) / T, T, 0, stream>>>(wq, wq_bf, DD * DD / 4);
  f32_to_bf16_vec4<<<(NKV * DD / 4 + T - 1) / T, T, 0, stream>>>(wk, wk_bf, NKV * DD / 4);
  f32_to_bf16_vec4<<<(NKV * DD / 4 + T - 1) / T, T, 0, stream>>>(wvp, wv_bf, NKV * DD / 4);
  f32_to_bf16_vec4<<<(DD * DD / 4 + T - 1) / T, T, 0, stream>>>(wo, wo_bf, DD * DD / 4);
  int ncache = BB * HKV * CACHE * HD;       // 1,048,576
  convert_cache_k<<<(ncache / 2 + T - 1) / T, T, 0, stream>>>(ck, Kt, ncache / 2);
  convert_cache_v<<<(ncache + T - 1) / T, T, 0, stream>>>(cv, Vt, ncache);
  // projections (bf16 WMMA, async-LDS staged)
  gemm_bf16_nt<DD, DD><<<dim3(DD / 128, BL / 128), T, 0, stream>>>(x_bf, wq_bf, Yq);
  gemm_bf16_nt<NKV, DD><<<dim3(NKV / 128, BL / 128), T, 0, stream>>>(x_bf, wk_bf, Yk);
  gemm_bf16_nt<NKV, DD><<<dim3(NKV / 128, BL / 128), T, 0, stream>>>(x_bf, wv_bf, Yv);
  // per-head RMS norms + V transpose-pack
  rmsnorm_q<<<BL * HH / 8, T, 0, stream>>>(Yq, Qb);
  rmsnorm_k<<<BL * HKV / 8, T, 0, stream>>>(Yk, Kt);
  scatter_v<<<(BL * NKV + T - 1) / T, T, 0, stream>>>(Yv, Vt, BL * NKV);
  gates_kernel<<<BL * HH / T, T, 0, stream>>>(x, wg, gates);
  // flash attention (Og aliases Yq — safe after rmsnorm_q)
  attn_kernel<<<(BB * HH * (LL / 16)) / 8, T, 0, stream>>>(Qb, Kt, Vt, gates, Og);
  // output projection
  gemm_bf16_nt<DD, DD><<<dim3(DD / 128, BL / 128), T, 0, stream>>>(Og, wo_bf, out);
}
